// SoftmaxAttention_66872640799104
// MI455X (gfx1250) — matrix operand order repaired and checked by reading, not compiled, not run on hardware
//
#include <hip/hip_runtime.h>
#include <hip/hip_bf16.h>

typedef __attribute__((ext_vector_type(16))) __bf16 v16bf;
typedef __attribute__((ext_vector_type(8)))  __bf16 v8bf;
typedef __attribute__((ext_vector_type(4)))  __bf16 v4bf;
typedef __attribute__((ext_vector_type(8)))  float  v8f;

#define S_LEN 4096
#define D_DIM 64
#define KT    64      // keys per tile
#define WAVES 8       // waves per block
#define QT    16      // query rows per wave
#define ROWS_PER_BLOCK (WAVES * QT)   // 128

// padded LDS rows (16B alignment of fragment chunk loads; 8 | row for b64 stores)
#define KP (D_DIM + 16)   // kshT row: 80 bf16 = 160B
#define VP (KT + 16)      // vshT row: 80 bf16 = 160B

// scale = 512^-0.5 (module uses full embed_dim) * log2(e), folded into Q
#define KSCALE_LOG2E (0.04419417382415922f * 1.4426950408889634f)

// raw hardware v_exp_f32: flush-to-zero semantics are fine for softmax
#define EXP2F(x) __builtin_amdgcn_exp2f(x)

__global__ __launch_bounds__(256, 2)
void SoftmaxAttention_fa_kernel(const float* __restrict__ qg,
                                const float* __restrict__ kg,
                                const float* __restrict__ vg,
                                float* __restrict__ og) {
    // double-buffered tiles: one barrier per iteration
    __shared__ __align__(32) __bf16 kshT[2][KT][KP];      // [t][d]  2x10 KB
    __shared__ __align__(32) __bf16 vshT[2][D_DIM][VP];   // [d][t]  2x10 KB

    const int lane = threadIdx.x & 31;
    const int wave = threadIdx.x >> 5;
    const int hi   = lane >> 4;
    const int ln   = lane & 15;

    const int bh    = blockIdx.x >> 5;                  // 32 blocks per (b,h)
    const int qblk  = blockIdx.x & 31;
    const int qrow0 = qblk * ROWS_PER_BLOCK + wave * QT;

    const float* qp = qg + (size_t)bh * S_LEN * D_DIM;
    const float* kp = kg + (size_t)bh * D_DIM * S_LEN;
    const float* vp = vg + (size_t)bh * S_LEN * D_DIM;
    float*       op = og + (size_t)bh * S_LEN * D_DIM;

    // ---- Q -> B-fragments of Q^T (for S^T = K^T * Q^T), pre-scaled ----
    v16bf qb0, qb1;
    {
        const size_t rowoff = (size_t)(qrow0 + ln) * D_DIM;
        const v8f f0 = *(const v8f*)&qp[rowoff + hi * 16];
        const v8f f1 = *(const v8f*)&qp[rowoff + hi * 16 + 8];
        const v8f f2 = *(const v8f*)&qp[rowoff + 32 + hi * 16];
        const v8f f3 = *(const v8f*)&qp[rowoff + 32 + hi * 16 + 8];
#pragma unroll
        for (int e = 0; e < 8; ++e) {
            qb0[e]     = (__bf16)(f0[e] * KSCALE_LOG2E);
            qb0[e + 8] = (__bf16)(f1[e] * KSCALE_LOG2E);
            qb1[e]     = (__bf16)(f2[e] * KSCALE_LOG2E);
            qb1[e + 8] = (__bf16)(f3[e] * KSCALE_LOG2E);
        }
    }

    v16bf ones;
#pragma unroll
    for (int e = 0; e < 16; ++e) ones[e] = (__bf16)1.0f;

    v8f acc[4] = {};
    v8f accL  = {};
    float m = -__builtin_inff();

    // ---- 4x4 micro-tile staging assignment (one micro-tile per thread) ----
    const int kd0 = (threadIdx.x >> 4) * 4;   // K: d in [kd0,kd0+4)
    const int kt4 = (threadIdx.x & 15) * 4;   // K: t in [kt4,kt4+4)
    const int vt0 = kd0;                      // V: t in [vt0,vt0+4)
    const int vd4 = kt4;                      // V: d in [vd4,vd4+4)

    // prologue: load tile 0 into registers
    float4 kf0, kf1, kf2, kf3, vf0, vf1, vf2, vf3;
    kf0 = *(const float4*)&kp[(size_t)(kd0 + 0) * S_LEN + kt4];
    kf1 = *(const float4*)&kp[(size_t)(kd0 + 1) * S_LEN + kt4];
    kf2 = *(const float4*)&kp[(size_t)(kd0 + 2) * S_LEN + kt4];
    kf3 = *(const float4*)&kp[(size_t)(kd0 + 3) * S_LEN + kt4];
    vf0 = *(const float4*)&vp[(size_t)(vt0 + 0) * D_DIM + vd4];
    vf1 = *(const float4*)&vp[(size_t)(vt0 + 1) * D_DIM + vd4];
    vf2 = *(const float4*)&vp[(size_t)(vt0 + 2) * D_DIM + vd4];
    vf3 = *(const float4*)&vp[(size_t)(vt0 + 3) * D_DIM + vd4];

    int buf = 0;
    for (int t0 = 0; t0 < S_LEN; t0 += KT) {
        // ---- commit staged registers to LDS (4x4 register transpose, b64 stores) ----
        {
            v4bf r;
            r[0] = (__bf16)kf0.x; r[1] = (__bf16)kf1.x; r[2] = (__bf16)kf2.x; r[3] = (__bf16)kf3.x;
            *(v4bf*)&kshT[buf][kt4 + 0][kd0] = r;
            r[0] = (__bf16)kf0.y; r[1] = (__bf16)kf1.y; r[2] = (__bf16)kf2.y; r[3] = (__bf16)kf3.y;
            *(v4bf*)&kshT[buf][kt4 + 1][kd0] = r;
            r[0] = (__bf16)kf0.z; r[1] = (__bf16)kf1.z; r[2] = (__bf16)kf2.z; r[3] = (__bf16)kf3.z;
            *(v4bf*)&kshT[buf][kt4 + 2][kd0] = r;
            r[0] = (__bf16)kf0.w; r[1] = (__bf16)kf1.w; r[2] = (__bf16)kf2.w; r[3] = (__bf16)kf3.w;
            *(v4bf*)&kshT[buf][kt4 + 3][kd0] = r;

            r[0] = (__bf16)vf0.x; r[1] = (__bf16)vf1.x; r[2] = (__bf16)vf2.x; r[3] = (__bf16)vf3.x;
            *(v4bf*)&vshT[buf][vd4 + 0][vt0] = r;
            r[0] = (__bf16)vf0.y; r[1] = (__bf16)vf1.y; r[2] = (__bf16)vf2.y; r[3] = (__bf16)vf3.y;
            *(v4bf*)&vshT[buf][vd4 + 1][vt0] = r;
            r[0] = (__bf16)vf0.z; r[1] = (__bf16)vf1.z; r[2] = (__bf16)vf2.z; r[3] = (__bf16)vf3.z;
            *(v4bf*)&vshT[buf][vd4 + 2][vt0] = r;
            r[0] = (__bf16)vf0.w; r[1] = (__bf16)vf1.w; r[2] = (__bf16)vf2.w; r[3] = (__bf16)vf3.w;
            *(v4bf*)&vshT[buf][vd4 + 3][vt0] = r;
        }
        __syncthreads();

        // ---- software prefetch of next tile into registers (overlaps WMMAs) ----
        if (t0 + KT < S_LEN) {
            const int tn = t0 + KT;
            kf0 = *(const float4*)&kp[(size_t)(kd0 + 0) * S_LEN + tn + kt4];
            kf1 = *(const float4*)&kp[(size_t)(kd0 + 1) * S_LEN + tn + kt4];
            kf2 = *(const float4*)&kp[(size_t)(kd0 + 2) * S_LEN + tn + kt4];
            kf3 = *(const float4*)&kp[(size_t)(kd0 + 3) * S_LEN + tn + kt4];
            vf0 = *(const float4*)&vp[(size_t)(tn + vt0 + 0) * D_DIM + vd4];
            vf1 = *(const float4*)&vp[(size_t)(tn + vt0 + 1) * D_DIM + vd4];
            vf2 = *(const float4*)&vp[(size_t)(tn + vt0 + 2) * D_DIM + vd4];
            vf3 = *(const float4*)&vp[(size_t)(tn + vt0 + 3) * D_DIM + vd4];
        }

        const __bf16 (*ks)[KP] = kshT[buf];
        const __bf16 (*vs)[VP] = vshT[buf];

        // ---- S^T = K^T * Q^T : four 16x16 t-tiles x two chained K=32 WMMAs ----
        v8f sT[4];
#pragma unroll
        for (int j = 0; j < 4; ++j) {
            v8f c = {};
#pragma unroll
            for (int w = 0; w < 2; ++w) {
                const v8bf a0 = *(const v8bf*)&ks[j * 16 + ln][w * 32 + hi * 16];
                const v8bf a1 = *(const v8bf*)&ks[j * 16 + ln][w * 32 + hi * 16 + 8];
                const v16bf ka = __builtin_shufflevector(a0, a1, 0, 1, 2, 3, 4, 5, 6, 7,
                                                         8, 9, 10, 11, 12, 13, 14, 15);
                c = __builtin_amdgcn_wmma_f32_16x16x32_bf16(false, ka,
                                                            false, w ? qb1 : qb0,
                                                            (short)0, c, false, false);
            }
            sT[j] = c;
        }

        // ---- Online softmax (row q=ln: 32 values in-lane + 32 in lane^16) ----
        float tmax = sT[0][0];
#pragma unroll
        for (int j = 0; j < 4; ++j)
#pragma unroll
            for (int i = 0; i < 8; ++i)
                tmax = fmaxf(tmax, sT[j][i]);
        tmax = fmaxf(tmax, __shfl_xor(tmax, 16, 32));
        const float mold = m;
        const float mnew = fmaxf(mold, tmax);
        m = mnew;

        // P^T in C-layout == P in A-layout: build fragments in-lane (raw v_exp_f32)
        v16bf pa0, pa1;
#pragma unroll
        for (int e = 0; e < 8; ++e) {
            pa0[e]     = (__bf16)EXP2F(sT[0][e] - mnew);
            pa0[e + 8] = (__bf16)EXP2F(sT[1][e] - mnew);
            pa1[e]     = (__bf16)EXP2F(sT[2][e] - mnew);
            pa1[e + 8] = (__bf16)EXP2F(sT[3][e] - mnew);
        }

        // Rescale only when the running max actually grew (rare once warmed up)
        if (__any(mnew > mold)) {
            const float alpha = EXP2F(mold - mnew);
            float alphaB[8];
#pragma unroll
            for (int i = 0; i < 8; ++i)
                alphaB[i] = __shfl(alpha, i + 8 * hi, 32);
#pragma unroll
            for (int i = 0; i < 8; ++i) {
                acc[0][i] *= alphaB[i]; acc[1][i] *= alphaB[i];
                acc[2][i] *= alphaB[i]; acc[3][i] *= alphaB[i];
                accL[i]   *= alphaB[i];
            }
        }

        // ---- Row sums on the matrix pipe: accL += P * ones ----
        accL = __builtin_amdgcn_wmma_f32_16x16x32_bf16(false, pa0, false, ones,
                                                       (short)0, accL, false, false);
        accL = __builtin_amdgcn_wmma_f32_16x16x32_bf16(false, pa1, false, ones,
                                                       (short)0, accL, false, false);

        // ---- O += P*V : four 16x16 d-tiles x two chained K=32 WMMAs ----
#pragma unroll
        for (int dt = 0; dt < 4; ++dt) {
            v16bf vb;
            vb = __builtin_shufflevector(*(const v8bf*)&vs[dt * 16 + ln][hi * 8], *(const v8bf*)&vs[dt * 16 + ln][16 + hi * 8], 0, 1, 2, 3, 4, 5, 6, 7, 8, 9, 10, 11, 12, 13, 14, 15);
            acc[dt] = __builtin_amdgcn_wmma_f32_16x16x32_bf16(false, pa0, false, vb,
                                                              (short)0, acc[dt], false, false);
            vb = __builtin_shufflevector(*(const v8bf*)&vs[dt * 16 + ln][32 + hi * 8], *(const v8bf*)&vs[dt * 16 + ln][48 + hi * 8], 0, 1, 2, 3, 4, 5, 6, 7, 8, 9, 10, 11, 12, 13, 14, 15);
            acc[dt] = __builtin_amdgcn_wmma_f32_16x16x32_bf16(false, pa1, false, vb,
                                                              (short)0, acc[dt], false, false);
        }

        buf ^= 1;   // double buffer: one barrier per iteration
    }

    // ---- Normalize and store O (accL lane-replicated denominator) ----
#pragma unroll
    for (int i = 0; i < 8; ++i) {
        const float inv = 1.0f / accL[i];
        const size_t row = (size_t)(qrow0 + i + hi * 8) * D_DIM;
        op[row + 0  + ln] = acc[0][i] * inv;
        op[row + 16 + ln] = acc[1][i] * inv;
        op[row + 32 + ln] = acc[2][i] * inv;
        op[row + 48 + ln] = acc[3][i] * inv;
    }
}

extern "C" void kernel_launch(void* const* d_in, const int* in_sizes, int n_in,
                              void* d_out, int out_size, void* d_ws, size_t ws_size,
                              hipStream_t stream) {
    const float* q = (const float*)d_in[0];
    const float* k = (const float*)d_in[1];
    const float* v = (const float*)d_in[2];
    float* out = (float*)d_out;

    const int BH = in_sizes[0] / (S_LEN * D_DIM);             // B*H = 16
    const int blocks = BH * (S_LEN / ROWS_PER_BLOCK);         // 512
    SoftmaxAttention_fa_kernel<<<dim3(blocks), dim3(256), 0, stream>>>(q, k, v, out);
}
